// SelectiveAttention_40192303956183
// MI455X (gfx1250) — matrix operand order repaired and checked by reading, not compiled, not run on hardware
//
#include <hip/hip_runtime.h>
#include <hip/hip_bf16.h>
#include <math.h>

#define L_SEQ   4096
#define D_MODEL 1024
#define N_HEADS 16
#define HEAD_DIM 64

typedef __attribute__((ext_vector_type(16))) _Float16 v16h;
typedef __attribute__((ext_vector_type(8)))  _Float16 v8h;
typedef __attribute__((ext_vector_type(8)))  float    v8f;

// ---------------------------------------------------------------------------
// WMMA helper: D = A(16x32 f16) x B(32x16 f16) + C(16x16 f32)
// ---------------------------------------------------------------------------
__device__ __forceinline__ v8f wmma_f16(v16h a, v16h b, v8f c) {
  return __builtin_amdgcn_wmma_f32_16x16x32_f16(
      /*neg_a=*/false, a, /*neg_b=*/false, b,
      /*c_mod=*/(short)0, c, /*reuse_a=*/false, /*reuse_b=*/false);
}

// A fragment (16x32 f16): lane m=lane&15 holds row m; half=lane>>4 selects K
// run {0..15} / {16..31} -> two contiguous 16B loads.
__device__ __forceinline__ v16h frag_A_h16(const _Float16* tile, int ld) {
  int lane = threadIdx.x & 31;
  int m    = lane & 15;
  int koff = (lane >> 4) * 16;
  const _Float16* p = tile + m * ld + koff;
  v8h lo = *(const v8h*)(p);
  v8h hi = *(const v8h*)(p + 8);
  v16h r;
#pragma unroll
  for (int i = 0; i < 8; ++i) { r[i] = lo[i]; r[i + 8] = hi[i]; }
  return r;
}

// B fragment (32x16 f16): lane holds column n=lane&15; K contiguous per lane.
// base[n*ld + k] = B[k][n].
__device__ __forceinline__ v16h frag_B_rows(const _Float16* tile, int ld) {
  int lane = threadIdx.x & 31;
  int n  = lane & 15;
  int kh = (lane >> 4) * 16;
  const _Float16* p = tile + n * ld + kh;
  v8h a = *(const v8h*)(p);
  v8h b = *(const v8h*)(p + 8);
  v16h r;
#pragma unroll
  for (int i = 0; i < 8; ++i) { r[i] = a[i]; r[i + 8] = b[i]; }
  return r;
}

// ---------------------------------------------------------------------------
// gfx1250 async global -> LDS copy (ASYNCcnt); 16B per active lane.
// VDST VGPR = LDS byte offset (low 32 bits of flat shared address).
// ---------------------------------------------------------------------------
__device__ __forceinline__ void gld_async_b128(void* lds_ptr, const void* gptr) {
  unsigned int ldsoff = (unsigned int)(unsigned long long)lds_ptr;
  asm volatile("global_load_async_to_lds_b128 %0, %1, off"
               :: "v"(ldsoff), "v"(gptr) : "memory");
}
#define S_WAIT_ASYNCCNT(n) asm volatile("s_wait_asynccnt " #n ::: "memory")

// ---------------------------------------------------------------------------
// Kernel 0a: fp32 -> f16 elementwise
// ---------------------------------------------------------------------------
__global__ void k_cvt(const float* __restrict__ src, _Float16* __restrict__ dst, int n) {
  int i = blockIdx.x * blockDim.x + threadIdx.x;
  if (i < n) dst[i] = (_Float16)src[i];
}

// Kernel 0b: fp32 [rows][cols] -> f16 transposed [cols][rows]
__global__ void k_cvt_T(const float* __restrict__ src, _Float16* __restrict__ dst,
                        int rows, int cols) {
  int i = blockIdx.x * blockDim.x + threadIdx.x;
  if (i < rows * cols) {
    int r = i / cols, c = i - r * cols;
    dst[c * rows + r] = (_Float16)src[i];
  }
}

// ---------------------------------------------------------------------------
// Kernel 1: gate = sigmoid(gelu(x@Wg1+bg1)@Wg2+bg2), 8 rows per 256-thread block
// ---------------------------------------------------------------------------
__global__ __launch_bounds__(256) void k_gate(
    const float* __restrict__ x, const float* __restrict__ Wg1,
    const float* __restrict__ bg1, const float* __restrict__ Wg2,
    const float* __restrict__ bg2, float* __restrict__ gate) {
  __shared__ float xs[8][D_MODEL];   // 32 KB
  __shared__ float red[8][256];      //  8 KB
  int t = threadIdx.x;
  int row0 = blockIdx.x * 8;
  for (int idx = t; idx < 8 * D_MODEL; idx += 256) {
    int r = idx >> 10, c = idx & (D_MODEL - 1);
    xs[r][c] = x[(row0 + r) * D_MODEL + c];
  }
  __syncthreads();
  float acc[8];
#pragma unroll
  for (int r = 0; r < 8; ++r) acc[r] = 0.f;
  for (int k = 0; k < D_MODEL; ++k) {
    float w = Wg1[k * 256 + t];
#pragma unroll
    for (int r = 0; r < 8; ++r) acc[r] = fmaf(xs[r][k], w, acc[r]);
  }
  float b1 = bg1[t];
  float w2 = Wg2[t];
#pragma unroll
  for (int r = 0; r < 8; ++r) {
    float h = acc[r] + b1;
    h = 0.5f * h * (1.0f + erff(h * 0.70710678f));   // exact GELU
    red[r][t] = h * w2;
  }
  __syncthreads();
  for (int s = 128; s > 0; s >>= 1) {
    if (t < s) {
#pragma unroll
      for (int r = 0; r < 8; ++r) red[r][t] += red[r][t + s];
    }
    __syncthreads();
  }
  if (t < 8) {
    float g = red[t][0] + bg2[0];
    gate[row0 + t] = 1.0f / (1.0f + __expf(-g));
  }
}

// ---------------------------------------------------------------------------
// Kernel 2: QKV projection. One wave per 16x16 tile of (4096 x 3072).
// Scatters into per-head Q[h][l][d], K[h][l][d], Vt[h][d][l] (f16).
// ---------------------------------------------------------------------------
__global__ __launch_bounds__(128) void k_qkv(
    const _Float16* __restrict__ xh, const _Float16* __restrict__ WT,
    _Float16* __restrict__ Q, _Float16* __restrict__ K, _Float16* __restrict__ Vt) {
  int wave = threadIdx.x >> 5;
  int row0 = blockIdx.x * 16;
  int c0   = (blockIdx.y * 4 + wave) * 16;
  int lane = threadIdx.x & 31;
  v8f acc;
#pragma unroll
  for (int i = 0; i < 8; ++i) acc[i] = 0.f;
  for (int k0 = 0; k0 < D_MODEL; k0 += 32) {
    v16h a = frag_A_h16(xh + row0 * D_MODEL + k0, D_MODEL);
    v16h b = frag_B_rows(WT + c0 * D_MODEL + k0, D_MODEL);
    acc = wmma_f16(a, b, acc);
  }
  int n = lane & 15, halfw = lane >> 4;
  int col = c0 + n;
  int sec = col >> 10;          // 0:Q 1:K 2:V
  int within = col & 1023;
  int h = within >> 6;
  int d = within & 63;
#pragma unroll
  for (int v = 0; v < 8; ++v) {
    int row = row0 + v + halfw * 8;
    _Float16 val = (_Float16)acc[v];
    if (sec == 0)      Q[(h * L_SEQ + row) * HEAD_DIM + d] = val;
    else if (sec == 1) K[(h * L_SEQ + row) * HEAD_DIM + d] = val;
    else               Vt[(h * HEAD_DIM + d) * L_SEQ + row] = val;
  }
}

// ---------------------------------------------------------------------------
// Kernel 3: causal flash attention. 4 waves/block = 4 adjacent 16-query tiles
// of one head sharing async double-buffered K/V LDS tiles (32-key steps).
// ---------------------------------------------------------------------------
__global__ __launch_bounds__(128) void k_attn(
    const _Float16* __restrict__ Q, const _Float16* __restrict__ K,
    const _Float16* __restrict__ Vt, _Float16* __restrict__ O) {
  __shared__ __align__(16) _Float16 Kt[2][32 * HEAD_DIM];   // 2 x 4 KB  [key][dim]
  __shared__ __align__(16) _Float16 Vtile[2][HEAD_DIM * 32];// 2 x 4 KB  [dim][key]
  __shared__ __align__(16) _Float16 Plds[4][16 * 32];       // 4 KB, per wave

  int h    = blockIdx.x;
  int Q0   = blockIdx.y * 64;
  int wave = threadIdx.x >> 5;
  int q0   = Q0 + wave * 16;
  int tid  = threadIdx.x;
  int lane = tid & 31;
  int n     = lane & 15;
  int halfw = lane >> 4;
  const _Float16* Qh  = Q  + (size_t)h * L_SEQ * HEAD_DIM;
  const _Float16* Kh  = K  + (size_t)h * L_SEQ * HEAD_DIM;
  const _Float16* Vth = Vt + (size_t)h * HEAD_DIM * L_SEQ;

  v16h aq0 = frag_A_h16(Qh + q0 * HEAD_DIM + 0,  HEAD_DIM);
  v16h aq1 = frag_A_h16(Qh + q0 * HEAD_DIM + 32, HEAD_DIM);

  float mrow[8], lrow[8];
  v8f accO[4];
#pragma unroll
  for (int v = 0; v < 8; ++v) { mrow[v] = -3.0e38f; lrow[v] = 0.f; }
#pragma unroll
  for (int g = 0; g < 4; ++g)
#pragma unroll
    for (int v = 0; v < 8; ++v) accO[g][v] = 0.f;

  const float scale = 0.125f;                       // 64^-0.5
  int nsteps_w   = (q0 + 16 + 31) >> 5;             // this wave's trip count
  int nsteps_max = (Q0 + 64 + 31) >> 5;             // block-uniform trip count

  // cooperative tile copy: 128 threads x 2 x 16B chunks cover each 4KB tile
  // K tile:  row = chunk>>3  (8 x 16B per 64-half row), off = (chunk&7)*8
  // V tile:  row = chunk>>2  (4 x 16B per 32-half row), off = (chunk&3)*8
  int kc0 = tid * 2, kc1 = tid * 2 + 1;

  // prologue: stage step 0 into buffer 0
  {
    int key0 = 0;
    gld_async_b128(&Kt[0][(kc0 >> 3) * HEAD_DIM + (kc0 & 7) * 8],
                   Kh + (key0 + (kc0 >> 3)) * HEAD_DIM + (kc0 & 7) * 8);
    gld_async_b128(&Kt[0][(kc1 >> 3) * HEAD_DIM + (kc1 & 7) * 8],
                   Kh + (key0 + (kc1 >> 3)) * HEAD_DIM + (kc1 & 7) * 8);
    gld_async_b128(&Vtile[0][(kc0 >> 2) * 32 + (kc0 & 3) * 8],
                   Vth + (size_t)(kc0 >> 2) * L_SEQ + key0 + (kc0 & 3) * 8);
    gld_async_b128(&Vtile[0][(kc1 >> 2) * 32 + (kc1 & 3) * 8],
                   Vth + (size_t)(kc1 >> 2) * L_SEQ + key0 + (kc1 & 3) * 8);
  }

  for (int t = 0; t < nsteps_max; ++t) {
    int key0 = t << 5;
    int buf  = t & 1;
    if (t + 1 < nsteps_max) {                       // stage t+1 into other buf
      int nb = (t + 1) & 1;
      int nk = (t + 1) << 5;
      gld_async_b128(&Kt[nb][(kc0 >> 3) * HEAD_DIM + (kc0 & 7) * 8],
                     Kh + (nk + (kc0 >> 3)) * HEAD_DIM + (kc0 & 7) * 8);
      gld_async_b128(&Kt[nb][(kc1 >> 3) * HEAD_DIM + (kc1 & 7) * 8],
                     Kh + (nk + (kc1 >> 3)) * HEAD_DIM + (kc1 & 7) * 8);
      gld_async_b128(&Vtile[nb][(kc0 >> 2) * 32 + (kc0 & 3) * 8],
                     Vth + (size_t)(kc0 >> 2) * L_SEQ + nk + (kc0 & 3) * 8);
      gld_async_b128(&Vtile[nb][(kc1 >> 2) * 32 + (kc1 & 3) * 8],
                     Vth + (size_t)(kc1 >> 2) * L_SEQ + nk + (kc1 & 3) * 8);
      S_WAIT_ASYNCCNT(0x4);                         // group t retired
    } else {
      S_WAIT_ASYNCCNT(0x0);
    }
    __syncthreads();                                // tiles visible block-wide

    if (t < nsteps_w) {
      // S = Q K^T from LDS K tile
      v8f s0, s1;
#pragma unroll
      for (int i = 0; i < 8; ++i) { s0[i] = 0.f; s1[i] = 0.f; }
      {
        v16h b = frag_B_rows(&Kt[buf][0], HEAD_DIM);
        s0 = wmma_f16(aq0, b, s0);
        b = frag_B_rows(&Kt[buf][32], HEAD_DIM);
        s0 = wmma_f16(aq1, b, s0);
        b = frag_B_rows(&Kt[buf][16 * HEAD_DIM], HEAD_DIM);
        s1 = wmma_f16(aq0, b, s1);
        b = frag_B_rows(&Kt[buf][16 * HEAD_DIM + 32], HEAD_DIM);
        s1 = wmma_f16(aq1, b, s1);
      }
      bool diag = (key0 + 31 >= q0);
#pragma unroll
      for (int v = 0; v < 8; ++v) {
        float e0 = s0[v] * scale;
        float e1 = s1[v] * scale;
        if (diag) {
          int row = q0 + v + halfw * 8;
          if (key0 + n > row)      e0 = -3.0e38f;
          if (key0 + 16 + n > row) e1 = -3.0e38f;
        }
        s0[v] = e0; s1[v] = e1;
      }
      // online softmax (16-lane halves hold one row each)
      float mnew[8], alpha[8];
#pragma unroll
      for (int v = 0; v < 8; ++v) {
        float mx = fmaxf(s0[v], s1[v]);
#pragma unroll
        for (int off = 8; off >= 1; off >>= 1)
          mx = fmaxf(mx, __shfl_xor(mx, off, 32));
        mnew[v]  = fmaxf(mrow[v], mx);
        alpha[v] = __expf(mrow[v] - mnew[v]);
        mrow[v]  = mnew[v];
      }
#pragma unroll
      for (int v = 0; v < 8; ++v) {
        float p0 = __expf(s0[v] - mnew[v]);
        float p1 = __expf(s1[v] - mnew[v]);
        float rs = p0 + p1;
#pragma unroll
        for (int off = 8; off >= 1; off >>= 1)
          rs += __shfl_xor(rs, off, 32);
        lrow[v] = lrow[v] * alpha[v] + rs;
        int row = v + halfw * 8;                    // C-layout -> [row][key]
        Plds[wave][row * 32 + n]      = (_Float16)p0;
        Plds[wave][row * 32 + 16 + n] = (_Float16)p1;
      }
#pragma unroll
      for (int g = 0; g < 4; ++g)
#pragma unroll
        for (int v = 0; v < 8; ++v) accO[g][v] *= alpha[v];
      v16h ap = frag_A_h16(&Plds[wave][0], 32);     // P as A-frag (K=32 keys)
#pragma unroll
      for (int g = 0; g < 4; ++g) {
        v16h bv = frag_B_rows(&Vtile[buf][g * 16 * 32], 32);
        accO[g] = wmma_f16(ap, bv, accO[g]);
      }
    }
    __syncthreads();                                // reads done before reuse
  }

#pragma unroll
  for (int v = 0; v < 8; ++v) lrow[v] = 1.0f / lrow[v];
#pragma unroll
  for (int g = 0; g < 4; ++g)
#pragma unroll
    for (int v = 0; v < 8; ++v) {
      int row = q0 + v + halfw * 8;
      int col = h * HEAD_DIM + g * 16 + n;
      O[row * D_MODEL + col] = (_Float16)(accO[g][v] * lrow[v]);
    }
}

// ---------------------------------------------------------------------------
// Kernel 4: out = gate * (O @ W_out) + (1-gate) * x   (fp32 output)
// ---------------------------------------------------------------------------
__global__ __launch_bounds__(128) void k_out(
    const _Float16* __restrict__ Oh, const _Float16* __restrict__ WoutT,
    const float* __restrict__ x, const float* __restrict__ gate,
    float* __restrict__ out) {
  int wave = threadIdx.x >> 5;
  int row0 = blockIdx.x * 16;
  int c0   = (blockIdx.y * 4 + wave) * 16;
  int lane = threadIdx.x & 31;
  v8f acc;
#pragma unroll
  for (int i = 0; i < 8; ++i) acc[i] = 0.f;
  for (int k0 = 0; k0 < D_MODEL; k0 += 32) {
    v16h a = frag_A_h16(Oh + row0 * D_MODEL + k0, D_MODEL);
    v16h b = frag_B_rows(WoutT + c0 * D_MODEL + k0, D_MODEL);
    acc = wmma_f16(a, b, acc);
  }
  int n = lane & 15, halfw = lane >> 4;
  int col = c0 + n;
#pragma unroll
  for (int v = 0; v < 8; ++v) {
    int row = row0 + v + halfw * 8;
    float g = gate[row];
    out[row * D_MODEL + col] = g * acc[v] + (1.0f - g) * x[row * D_MODEL + col];
  }
}

// ---------------------------------------------------------------------------
extern "C" void kernel_launch(void* const* d_in, const int* in_sizes, int n_in,
                              void* d_out, int out_size, void* d_ws, size_t ws_size,
                              hipStream_t stream) {
  (void)in_sizes; (void)n_in; (void)out_size; (void)ws_size;
  const float* x    = (const float*)d_in[0];
  const float* Wqkv = (const float*)d_in[1];
  const float* Wg1  = (const float*)d_in[2];
  const float* bg1  = (const float*)d_in[3];
  const float* Wg2  = (const float*)d_in[4];
  const float* bg2  = (const float*)d_in[5];
  const float* Wout = (const float*)d_in[6];
  float* out  = (float*)d_out;
  float* gate = out + (size_t)L_SEQ * D_MODEL;   // second tuple output

  char* ws = (char*)d_ws;                        // 48 MB used
  _Float16* xh    = (_Float16*)(ws);                           // 8 MB
  _Float16* WqkvT = (_Float16*)(ws + ( 8ull << 20));           // 6 MB
  _Float16* WoutT = (_Float16*)(ws + (14ull << 20));           // 2 MB
  _Float16* Qb    = (_Float16*)(ws + (16ull << 20));           // 8 MB
  _Float16* Kb    = (_Float16*)(ws + (24ull << 20));           // 8 MB
  _Float16* Vtb   = (_Float16*)(ws + (32ull << 20));           // 8 MB
  _Float16* Ob    = (_Float16*)(ws + (40ull << 20));           // 8 MB

  k_cvt  <<<(L_SEQ * D_MODEL + 255) / 256, 256, 0, stream>>>(x, xh, L_SEQ * D_MODEL);
  k_cvt_T<<<(D_MODEL * 3 * D_MODEL + 255) / 256, 256, 0, stream>>>(Wqkv, WqkvT, D_MODEL, 3 * D_MODEL);
  k_cvt_T<<<(D_MODEL * D_MODEL + 255) / 256, 256, 0, stream>>>(Wout, WoutT, D_MODEL, D_MODEL);
  k_gate <<<L_SEQ / 8, 256, 0, stream>>>(x, Wg1, bg1, Wg2, bg2, gate);
  k_qkv  <<<dim3(L_SEQ / 16, (3 * D_MODEL) / 64), 128, 0, stream>>>(xh, WqkvT, Qb, Kb, Vtb);
  k_attn <<<dim3(N_HEADS, L_SEQ / 64), 128, 0, stream>>>(Qb, Kb, Vtb, Ob);
  k_out  <<<dim3(L_SEQ / 16, D_MODEL / 64), 128, 0, stream>>>(Ob, WoutT, x, gate, out);
}
